// MyRNN_84000970375266
// MI455X (gfx1250) — matrix operand order repaired and checked by reading, not compiled, not run on hardware
//
#include <hip/hip_runtime.h>
#include <math.h>

typedef _Float16 v8h  __attribute__((ext_vector_type(8)));
typedef _Float16 v16h __attribute__((ext_vector_type(16)));
typedef float    v8f  __attribute__((ext_vector_type(8)));

#define SEQ   80
#define EMB   128
#define UNITS 128
#define BATCH 4096

#define WAVES   2
#define THREADS (WAVES * 32)
#define BTILE   (WAVES * 16)   // 32 batch rows per block -> 128 blocks

// ---- hardware tanh (CDNA5 TRANS op) with safe fallback ----------------------
__device__ __forceinline__ float fast_tanh(float x) {
#if __has_builtin(__builtin_amdgcn_tanhf)
  return __builtin_amdgcn_tanhf(x);
#elif __has_builtin(__builtin_amdgcn_tanh_f32)
  return __builtin_amdgcn_tanh_f32(x);
#else
  return tanhf(x);
#endif
}

// ---- WMMA wrapper: D = A(16x32 f16) * B(32x16 f16) + C(f32) -----------------
__device__ __forceinline__ v8f wmma_f16(v16h a, v16h b, v8f c) {
  return __builtin_amdgcn_wmma_f32_16x16x32_f16(
      /*neg_a=*/false, a, /*neg_b=*/false, b,
      /*c_mod=*/(short)0, c, /*reuse_a=*/false, /*reuse_b=*/false);
}

__device__ __forceinline__ v16h comb(v8h lo, v8h hi) {
  v16h f;
#pragma unroll
  for (int i = 0; i < 8; ++i) { f[i] = lo[i]; f[i + 8] = hi[i]; }
  return f;
}

// A fragment (16x32, f16) from a row-major f16 LDS matrix, row stride 128.
// ISA A layout: lane L holds row M=L%16; lanes<16 get K {0..7,16..23},
// lanes>=16 get K {8..15,24..31}.
__device__ __forceinline__ v16h load_afrag(const _Float16* base, int lane, int kt) {
  const _Float16* p = base + (lane & 15) * UNITS + kt * 32 + ((lane >> 4) << 3);
  return comb(*(const v8h*)p, *(const v8h*)(p + 16));
}

// B fragment (32x16, f16) from W^T stored row-major [n][k], row stride 128.
// ISA B layout: lane L holds column N=L%16; lanes<16 K {0..7,16..23}, lanes>=16 K {8..15,24..31}.
__device__ __forceinline__ v16h load_bfrag(const _Float16* wt, int lane, int nt, int kt) {
  const _Float16* p = wt + (nt * 16 + (lane & 15)) * UNITS + kt * 32 + ((lane >> 4) << 3);
  return comb(*(const v8h*)p, *(const v8h*)(p + 16));
}

// A fragment gathered directly from a global f32 row (embedding), f32->f16.
__device__ __forceinline__ v16h load_afrag_f32(const float* __restrict__ row, int lane, int kt) {
  const float* p = row + kt * 32 + ((lane >> 4) << 3);
  v16h f;
#pragma unroll
  for (int i = 0; i < 8; ++i) { f[i] = (_Float16)p[i]; f[i + 8] = (_Float16)p[i + 16]; }
  return f;
}

// Store a 16x16 f32 accumulator tile (D layout) through tanh into a row-major
// f16 LDS strip. VGPR r: lanes<16 -> M=r, N=lane; lanes>=16 -> M=r+8, N=lane-16.
__device__ __forceinline__ void store_htile(_Float16* strip, int lane, int nt, v8f acc) {
  const int n  = nt * 16 + (lane & 15);
  const int mb = (lane >> 4) << 3;
#pragma unroll
  for (int r = 0; r < 8; ++r)
    strip[(mb + r) * UNITS + n] = (_Float16)fast_tanh(acc[r]);
}

__device__ __forceinline__ void load_wT(const float* __restrict__ W, _Float16* WT, int tid) {
  for (int i = tid; i < UNITS * UNITS; i += THREADS) {
    const int k = i >> 7, n = i & 127;
    WT[n * UNITS + k] = (_Float16)W[i];   // store transposed: WT[n][k] = W[k][n]
  }
}

__global__ void __launch_bounds__(THREADS)
rnn2_wmma_kernel(const int*   __restrict__ inputs,  // [B, SEQ]
                 const float* __restrict__ emb,     // [VOCAB, EMB]
                 const float* __restrict__ Wx0, const float* __restrict__ Wh0,
                 const float* __restrict__ b0,
                 const float* __restrict__ Wx1, const float* __restrict__ Wh1,
                 const float* __restrict__ b1,
                 const float* __restrict__ Wout, const float* __restrict__ bout,
                 float* __restrict__ out)           // [B]
{
  extern __shared__ __align__(16) _Float16 lds[];
  _Float16* sWx0 = lds;                       // 128x128 each, transposed
  _Float16* sWh0 = sWx0 + UNITS * UNITS;
  _Float16* sWx1 = sWh0 + UNITS * UNITS;
  _Float16* sWh1 = sWx1 + UNITS * UNITS;
  _Float16* sH0  = sWh1 + UNITS * UNITS;      // BTILE x 128
  _Float16* sH1  = sH0  + BTILE * UNITS;      // BTILE x 128

  const int tid  = threadIdx.x;
  const int lane = tid & 31;
  const int wave = tid >> 5;

  // One-time: weights -> LDS (f16, transposed), zero hidden state.
  load_wT(Wx0, sWx0, tid);
  load_wT(Wh0, sWh0, tid);
  load_wT(Wx1, sWx1, tid);
  load_wT(Wh1, sWh1, tid);
  for (int i = tid; i < BTILE * UNITS; i += THREADS) {
    sH0[i] = (_Float16)0.f; sH1[i] = (_Float16)0.f;
  }
  __syncthreads();

  // Each wave owns an independent 16-row batch strip: no barriers below.
  _Float16* h0s = sH0 + wave * 16 * UNITS;
  _Float16* h1s = sH1 + wave * 16 * UNITS;
  const int rowb  = blockIdx.x * BTILE + wave * 16;
  const int myrow = rowb + (lane & 15);

  float b0v[8], b1v[8];
#pragma unroll
  for (int nt = 0; nt < 8; ++nt) {
    b0v[nt] = b0[nt * 16 + (lane & 15)];
    b1v[nt] = b1[nt * 16 + (lane & 15)];
  }

  for (int t = 0; t < SEQ; ++t) {
    const int tok = inputs[myrow * SEQ + t];
    const float* xrow = emb + (long)tok * EMB;

    // Layer 0: h0 = tanh(x @ Wx0 + h0 @ Wh0 + b0)
    v16h ax[4], ah[4];
#pragma unroll
    for (int kt = 0; kt < 4; ++kt) ax[kt] = load_afrag_f32(xrow, lane, kt);
#pragma unroll
    for (int kt = 0; kt < 4; ++kt) ah[kt] = load_afrag(h0s, lane, kt);

#pragma unroll
    for (int nt = 0; nt < 8; ++nt) {
      v8f acc;
#pragma unroll
      for (int r = 0; r < 8; ++r) acc[r] = b0v[nt];
#pragma unroll
      for (int kt = 0; kt < 4; ++kt) acc = wmma_f16(ax[kt], load_bfrag(sWx0, lane, nt, kt), acc);
#pragma unroll
      for (int kt = 0; kt < 4; ++kt) acc = wmma_f16(ah[kt], load_bfrag(sWh0, lane, nt, kt), acc);
      store_htile(h0s, lane, nt, acc);
    }

    // Layer 1: h1 = tanh(h0_new @ Wx1 + h1 @ Wh1 + b1)
    v16h a0[4], a1[4];
#pragma unroll
    for (int kt = 0; kt < 4; ++kt) a0[kt] = load_afrag(h0s, lane, kt);
#pragma unroll
    for (int kt = 0; kt < 4; ++kt) a1[kt] = load_afrag(h1s, lane, kt);

#pragma unroll
    for (int nt = 0; nt < 8; ++nt) {
      v8f acc;
#pragma unroll
      for (int r = 0; r < 8; ++r) acc[r] = b1v[nt];
#pragma unroll
      for (int kt = 0; kt < 4; ++kt) acc = wmma_f16(a0[kt], load_bfrag(sWx1, lane, nt, kt), acc);
#pragma unroll
      for (int kt = 0; kt < 4; ++kt) acc = wmma_f16(a1[kt], load_bfrag(sWh1, lane, nt, kt), acc);
      store_htile(h1s, lane, nt, acc);
    }
  }

  // Head: prob = sigmoid(h1 @ Wout + bout). 16 rows per wave, one lane each.
  if (lane < 16) {
    const _Float16* hr = h1s + lane * UNITS;
    float s = bout[0];
#pragma unroll 8
    for (int k = 0; k < UNITS; ++k) s += (float)hr[k] * Wout[k];
    out[rowb + lane] = 1.0f / (1.0f + expf(-s));
  }
}

extern "C" void kernel_launch(void* const* d_in, const int* in_sizes, int n_in,
                              void* d_out, int out_size, void* d_ws, size_t ws_size,
                              hipStream_t stream) {
  (void)in_sizes; (void)n_in; (void)d_ws; (void)ws_size; (void)out_size;
  const int*   inputs = (const int*)  d_in[0];
  const float* emb    = (const float*)d_in[1];
  const float* Wx0    = (const float*)d_in[2];
  const float* Wh0    = (const float*)d_in[3];
  const float* b0     = (const float*)d_in[4];
  const float* Wx1    = (const float*)d_in[5];
  const float* Wh1    = (const float*)d_in[6];
  const float* b1     = (const float*)d_in[7];
  const float* Wout   = (const float*)d_in[8];
  const float* bout   = (const float*)d_in[9];
  float*       out    = (float*)d_out;

  const size_t smem = (size_t)(4 * UNITS * UNITS + 2 * BTILE * UNITS) * sizeof(_Float16); // 144 KB
  dim3 grid(BATCH / BTILE);   // 128 blocks x 2 waves = 256 independent 16-row strips
  rnn2_wmma_kernel<<<grid, THREADS, smem, stream>>>(
      inputs, emb, Wx0, Wh0, b0, Wx1, Wh1, b1, Wout, bout, out);
}
